// EnhancedGAT_85143431676095
// MI455X (gfx1250) — matrix operand order repaired and checked by reading, not compiled, not run on hardware
//
#include <hip/hip_runtime.h>
#include <hip/hip_bf16.h>

typedef __bf16 bf16_t;
typedef __attribute__((ext_vector_type(16))) __bf16 v16bf;
typedef __attribute__((ext_vector_type(8)))  __bf16 v8bf;
typedef __attribute__((ext_vector_type(8)))  float   v8f;

__device__ __forceinline__ v8f wmma_bf16(v16bf a, v16bf b, v8f c) {
    return __builtin_amdgcn_wmma_f32_16x16x32_bf16(false, a, false, b, (short)0, c, false, false);
}
__device__ __forceinline__ v16bf cat8(v8bf lo, v8bf hi) {
    return __builtin_shufflevector(lo, hi, 0,1,2,3,4,5,6,7,8,9,10,11,12,13,14,15);
}

// ---------------------------------------------------------------------------
// Utility kernels
// ---------------------------------------------------------------------------
__global__ void fill_f32(float* p, float v, long long n) {
    long long i = (long long)blockIdx.x * blockDim.x + threadIdx.x;
    if (i < n) p[i] = v;
}
__global__ void fill_u32(unsigned* p, unsigned v, long long n) {
    long long i = (long long)blockIdx.x * blockDim.x + threadIdx.x;
    if (i < n) p[i] = v;
}
__global__ void cvt_bf16(const float* __restrict__ s, bf16_t* __restrict__ d, long long n) {
    long long i = (long long)blockIdx.x * blockDim.x + threadIdx.x;
    if (i < n) d[i] = (bf16_t)s[i];
}
__global__ void bias_add_rows(float* x, const float* __restrict__ b, long long rows, int cols) {
    long long i = (long long)blockIdx.x * blockDim.x + threadIdx.x;
    long long tot = rows * cols;
    if (i < tot) x[i] += b[i % cols];
}
// out[n,c] = bias[c] + mean_h in[n, h*C + c]
__global__ void head_mean(const float* __restrict__ in, const float* __restrict__ bias,
                          float* __restrict__ out, long long rows, int H, int C) {
    long long i = (long long)blockIdx.x * blockDim.x + threadIdx.x;
    long long tot = rows * C;
    if (i >= tot) return;
    long long n = i / C; int c = (int)(i % C);
    float s = 0.f;
    for (int h = 0; h < H; ++h) s += in[n * (long long)H * C + (long long)h * C + c];
    out[i] = s / (float)H + bias[c];
}

// ---------------------------------------------------------------------------
// WMMA GEMM (fast path): C[M,N] = act( A@B + bias + (accum?C:0) ), bf16 in / f32 out
// REQUIREMENTS (guaranteed by caller): M % 128 == 0, N % 128 == 0, K % 32 == 0.
// Block = 256 thr = 8 waves; block tile 128(M) x 128(N); wave tile 32 x 64.
// 8 v_wmma_f32_16x16x32_bf16 per K-step per wave.  act: 0=none, 1=ELU
// ---------------------------------------------------------------------------
__global__ __launch_bounds__(256) void gemm_bf16_wmma(
    const bf16_t* __restrict__ A, const bf16_t* __restrict__ B,
    const float* __restrict__ bias, float* __restrict__ C,
    int M, int N, int K, int act, int accum)
{
    const int lane = threadIdx.x & 31;
    const int wid  = threadIdx.x >> 5;
    const int wm   = wid & 3;          // 4 waves along M
    const int wn   = wid >> 2;         // 2 waves along N
    const int m0 = blockIdx.y * 128 + wm * 32;
    const int n0 = blockIdx.x * 128 + wn * 64;

    // A-fragment lane mapping (16-bit A 16x32, ISA 7.12.2):
    //   lane L (L<16): row m0+L,   halves 0..7 -> K=k0..k0+7,  8..15 -> k0+16..k0+23
    //   lane L (>=16): row m0+L-16,halves 0..7 -> k0+8..k0+15, 8..15 -> k0+24..k0+31
    const int arow  = lane & 15;
    const int kbase = (lane >> 4) * 8;
    const bf16_t* a0p = A + (size_t)(m0 +      arow) * K + kbase;
    const bf16_t* a1p = A + (size_t)(m0 + 16 + arow) * K + kbase;
    // B-fragment lane mapping: lane L holds col N = n+L%16, elements -> same K order as A
    const bf16_t* bp  = B + (size_t)((lane >> 4) * 8) * N + n0 + (lane & 15);

    v8f acc[2][4] = {};

    for (int k0 = 0; k0 < K; k0 += 32) {
        const v16bf a0 = cat8(*(const v8bf*)a0p, *(const v8bf*)(a0p + 16));
        const v16bf a1 = cat8(*(const v8bf*)a1p, *(const v8bf*)(a1p + 16));
        a0p += 32; a1p += 32;
#pragma unroll
        for (int j = 0; j < 4; ++j) {
            v16bf b;
#pragma unroll
            for (int s = 0; s < 16; ++s) b[s] = bp[(size_t)((s & 7) + (s >> 3) * 16) * N + j * 16];
            acc[0][j] = wmma_bf16(a0, b, acc[0][j]);
            acc[1][j] = wmma_bf16(a1, b, acc[1][j]);
        }
        bp += (size_t)32 * N;
    }

    // f32 C/D 16x16 layout: lane -> col n = lane%16; VGPR v -> row m = v + 8*(lane/16)
    const int mrow = m0 + 8 * (lane >> 4);
    const int ncol = n0 + (lane & 15);
#pragma unroll
    for (int i = 0; i < 2; ++i) {
#pragma unroll
        for (int j = 0; j < 4; ++j) {
            const int n = ncol + j * 16;
            const float bv = bias ? bias[n] : 0.0f;
#pragma unroll
            for (int v = 0; v < 8; ++v) {
                const int m = mrow + i * 16 + v;
                const size_t off = (size_t)m * N + n;
                float val = acc[i][j][v] + bv;
                if (accum) val += C[off];
                if (act == 1) val = val > 0.f ? val : __expf(val) - 1.f;
                C[off] = val;
            }
        }
    }
}

// Tiny final layer: out[n,j] = b[j] + sum_k X[n,k]*W[k,j]   (N=3, K=256; f32)
__global__ void mlp3_kernel(const float* __restrict__ X, const float* __restrict__ W,
                            const float* __restrict__ b, float* __restrict__ out, int Nn)
{
    int t = blockIdx.x * blockDim.x + threadIdx.x;
    if (t >= Nn * 3) return;
    const int n = t / 3, j = t % 3;
    const float* xp = X + (size_t)n * 256;
    float s = b[j];
#pragma unroll 8
    for (int k = 0; k < 256; ++k) s += xp[k] * W[k * 3 + j];
    out[t] = s;
}

// ---------------------------------------------------------------------------
// GATv2 edge-phase kernels
// ---------------------------------------------------------------------------
__device__ __forceinline__ unsigned ordenc(float f) {
    unsigned u = __float_as_uint(f);
    return (u & 0x80000000u) ? ~u : (u | 0x80000000u);
}
__device__ __forceinline__ float orddec(unsigned u) {
    return (u & 0x80000000u) ? __uint_as_float(u & 0x7fffffffu) : __uint_as_float(~u);
}

// One wave per (edge, head): logit = sum_c att[h,c] * lrelu0.2( xl[src]+xr[dst]+ef )
// ef computed on the fly from edge_attr[e,0:16] @ We[16, H*C].
__global__ __launch_bounds__(256) void gat_logits(
    const float* __restrict__ xl, const float* __restrict__ xr,
    const float* __restrict__ ea, const float* __restrict__ We,
    const float* __restrict__ att, const int* __restrict__ ei,
    int E, int H, int C, float* __restrict__ logits, unsigned* __restrict__ lmax_enc)
{
    const int lane = threadIdx.x & 31;
    const long long wv = ((long long)blockIdx.x * blockDim.x + threadIdx.x) >> 5;
    const long long total = (long long)E * H;
    if (wv >= total) return;
    const int e  = (int)(wv / H);
    const int hd = (int)(wv % H);
    const int src = ei[e];
    const int dst = ei[E + e];
    const int HC = H * C;

    float eal[16];
#pragma unroll
    for (int k = 0; k < 16; ++k) eal[k] = ea[(size_t)e * 16 + k];

    const float* pl = xl + (size_t)src * HC + (size_t)hd * C;
    const float* pr = xr + (size_t)dst * HC + (size_t)hd * C;
    const float* pa = att + (size_t)hd * C;
    const float* pw = We + (size_t)hd * C;

    float sum = 0.f;
    for (int c = lane; c < C; c += 32) {
        float ef = 0.f;
#pragma unroll
        for (int k = 0; k < 16; ++k) ef += eal[k] * pw[(size_t)k * HC + c];
        float m = pl[c] + pr[c] + ef;
        m = (m > 0.f) ? m : 0.2f * m;
        sum += pa[c] * m;
    }
#pragma unroll
    for (int off = 16; off >= 1; off >>= 1) sum += __shfl_xor(sum, off, 32);
    if (lane == 0) {
        logits[wv] = sum;
        atomicMax(&lmax_enc[(size_t)dst * H + hd], ordenc(sum));
    }
}

__global__ void lmax_decode(const unsigned* __restrict__ enc, float* __restrict__ out, long long n) {
    long long i = (long long)blockIdx.x * blockDim.x + threadIdx.x;
    if (i >= n) return;
    float v = orddec(enc[i]);
    out[i] = isfinite(v) ? v : 0.0f;
}

__global__ void gat_expden(const float* __restrict__ logits, const float* __restrict__ lmaxf,
                           const int* __restrict__ ei, int E, int H,
                           float* __restrict__ ex, float* __restrict__ den)
{
    long long t = (long long)blockIdx.x * blockDim.x + threadIdx.x;
    if (t >= (long long)E * H) return;
    const int e  = (int)(t / H);
    const int hd = (int)(t % H);
    const int dst = ei[E + e];
    float v = __expf(logits[t] - lmaxf[(size_t)dst * H + hd]);
    ex[t] = v;
    atomicAdd(&den[(size_t)dst * H + hd], v);
}

__global__ void gat_scatter(const float* __restrict__ xl, const float* __restrict__ ex,
                            const float* __restrict__ den, const int* __restrict__ ei,
                            int E, int H, int C, float* __restrict__ out)
{
    long long idx = (long long)blockIdx.x * blockDim.x + threadIdx.x;
    const int HC = H * C;
    const long long tot = (long long)E * HC;
    if (idx >= tot) return;
    const int e = (int)(idx / HC);
    const int r = (int)(idx % HC);
    const int hd = r / C;
    const int src = ei[e];
    const int dst = ei[E + e];
    const float alpha = ex[(size_t)e * H + hd] / (den[(size_t)dst * H + hd] + 1e-16f);
    atomicAdd(&out[(size_t)dst * HC + r], alpha * xl[(size_t)src * HC + r]);
}

// ---------------------------------------------------------------------------
// BatchNorm (node-dim statistics) and LayerNorm
// ---------------------------------------------------------------------------
__global__ void bn_stats(const float* __restrict__ x, const float* __restrict__ g,
                         const float* __restrict__ b, int rows, int cols,
                         float* __restrict__ scale, float* __restrict__ shift)
{
    int c = blockIdx.x * blockDim.x + threadIdx.x;
    if (c >= cols) return;
    float s = 0.f, s2 = 0.f;
    for (int n = 0; n < rows; ++n) {
        float v = x[(size_t)n * cols + c];
        s += v; s2 += v * v;
    }
    float mu = s / rows;
    float var = s2 / rows - mu * mu;
    float sc = g[c] * rsqrtf(var + 1e-5f);
    scale[c] = sc;
    shift[c] = b[c] - mu * sc;
}

// act: 0=none, 1=leaky0.1, 2=elu  (in/out may alias)
__global__ void bn_apply(const float* __restrict__ in, float* __restrict__ out,
                         const float* __restrict__ sc, const float* __restrict__ sh,
                         long long rows, int cols, int act)
{
    long long i = (long long)blockIdx.x * blockDim.x + threadIdx.x;
    long long tot = rows * cols;
    if (i >= tot) return;
    int c = (int)(i % cols);
    float v = sc[c] * in[i] + sh[c];
    if (act == 1) v = v > 0.f ? v : 0.1f * v;
    else if (act == 2) v = v > 0.f ? v : __expf(v) - 1.f;
    out[i] = v;
}

// LayerNorm over last dim followed by ELU; one block per row.
__global__ __launch_bounds__(256) void ln_elu(float* __restrict__ x,
                                              const float* __restrict__ g,
                                              const float* __restrict__ b, int cols)
{
    __shared__ float red0[8], red1[8];
    const int row = blockIdx.x;
    const int t = threadIdx.x;
    float s = 0.f, s2 = 0.f;
    for (int c = t; c < cols; c += 256) {
        float v = x[(size_t)row * cols + c];
        s += v; s2 += v * v;
    }
#pragma unroll
    for (int off = 16; off >= 1; off >>= 1) {
        s  += __shfl_xor(s,  off, 32);
        s2 += __shfl_xor(s2, off, 32);
    }
    const int lane = t & 31, w = t >> 5;
    if (lane == 0) { red0[w] = s; red1[w] = s2; }
    __syncthreads();
    if (t == 0) {
        float a = 0.f, a2 = 0.f;
        for (int i = 0; i < 8; ++i) { a += red0[i]; a2 += red1[i]; }
        red0[0] = a; red1[0] = a2;
    }
    __syncthreads();
    const float mu = red0[0] / cols;
    const float var = red1[0] / cols - mu * mu;
    const float r = rsqrtf(var + 1e-5f);
    for (int c = t; c < cols; c += 256) {
        size_t o = (size_t)row * cols + c;
        float v = g[c] * (x[o] - mu) * r + b[c];
        x[o] = v > 0.f ? v : __expf(v) - 1.f;
    }
}

// ---------------------------------------------------------------------------
// Host side
// ---------------------------------------------------------------------------
static inline int cdiv(long long a, long long b) { return (int)((a + b - 1) / b); }

static void gemm(hipStream_t s, const bf16_t* A, const bf16_t* B, const float* bias,
                 float* C, int M, int N, int K, int act, int accum)
{
    dim3 g(N / 128, M / 128);   // exact by construction
    gemm_bf16_wmma<<<g, 256, 0, s>>>(A, B, bias, C, M, N, K, act, accum);
}
static void cvt(hipStream_t s, const float* src, bf16_t* dst, long long n) {
    cvt_bf16<<<cdiv(n, 256), 256, 0, s>>>(src, dst, n);
}

static void run_gat_edges(hipStream_t s, const float* xl, const float* xr,
                          const float* ea, const float* We, const float* att,
                          const int* ei, int Nn, int E, int H, int C,
                          float* logits, float* ex, unsigned* lmax_enc,
                          float* lmaxf, float* den, float* out /* zeroed [Nn,H*C] */)
{
    const long long NH = (long long)Nn * H;
    const long long EH = (long long)E * H;
    fill_u32<<<cdiv(NH, 256), 256, 0, s>>>(lmax_enc, 0x007FFFFFu /* enc(-inf) */, NH);
    fill_f32<<<cdiv(NH, 256), 256, 0, s>>>(den, 0.f, NH);
    gat_logits<<<cdiv(EH * 32, 256), 256, 0, s>>>(xl, xr, ea, We, att, ei, E, H, C, logits, lmax_enc);
    lmax_decode<<<cdiv(NH, 256), 256, 0, s>>>(lmax_enc, lmaxf, NH);
    gat_expden<<<cdiv(EH, 256), 256, 0, s>>>(logits, lmaxf, ei, E, H, ex, den);
    const long long tot = EH * C;
    gat_scatter<<<cdiv(tot, 256), 256, 0, s>>>(xl, ex, den, ei, E, H, C, out);
}

extern "C" void kernel_launch(void* const* d_in, const int* in_sizes, int n_in,
                              void* d_out, int out_size, void* d_ws, size_t ws_size,
                              hipStream_t stream)
{
    // ---- model dimensions ----
    const int Nn = in_sizes[0] / 64;     // 8192 nodes
    const int E  = in_sizes[1] / 16;     // 65536 edges
    const int HID = 512;

    // ---- inputs (setup_inputs() dict order, depth-first) ----
    const float* x    = (const float*)d_in[0];
    const float* ea   = (const float*)d_in[1];
    const float* npW  = (const float*)d_in[2];
    const float* npB  = (const float*)d_in[3];
    const float* c1Wl = (const float*)d_in[4];
    const float* c1bl = (const float*)d_in[5];
    const float* c1Wr = (const float*)d_in[6];
    const float* c1br = (const float*)d_in[7];
    const float* c1We = (const float*)d_in[8];
    const float* c1at = (const float*)d_in[9];
    const float* c1bs = (const float*)d_in[10];
    const float* c2Wl = (const float*)d_in[11];
    const float* c2bl = (const float*)d_in[12];
    const float* c2Wr = (const float*)d_in[13];
    const float* c2br = (const float*)d_in[14];
    const float* c2We = (const float*)d_in[15];
    const float* c2at = (const float*)d_in[16];
    const float* c2bs = (const float*)d_in[17];
    const float* c3Wl = (const float*)d_in[18];
    const float* c3bl = (const float*)d_in[19];
    const float* c3Wr = (const float*)d_in[20];
    const float* c3br = (const float*)d_in[21];
    const float* c3We = (const float*)d_in[22];
    const float* c3at = (const float*)d_in[23];
    const float* c3bs = (const float*)d_in[24];
    const float* r1W  = (const float*)d_in[25];
    const float* r1b  = (const float*)d_in[26];
    const float* r2W  = (const float*)d_in[27];
    const float* r2b  = (const float*)d_in[28];
    const float* bn1g = (const float*)d_in[29];
    const float* bn1b = (const float*)d_in[30];
    const float* bn2g = (const float*)d_in[31];
    const float* bn2b = (const float*)d_in[32];
    const float* bn3g = (const float*)d_in[33];
    const float* bn3b = (const float*)d_in[34];
    const float* m1W  = (const float*)d_in[35];
    const float* m1b  = (const float*)d_in[36];
    const float* lng  = (const float*)d_in[37];
    const float* lnb  = (const float*)d_in[38];
    const float* m2W  = (const float*)d_in[39];
    const float* m2b  = (const float*)d_in[40];
    const float* m3W  = (const float*)d_in[41];
    const float* m3b  = (const float*)d_in[42];
    const int*   ei   = (const int*)d_in[43];

    float* out = (float*)d_out;

    // ---- workspace carve-out ----
    size_t off = 0;
    char* base = (char*)d_ws;
    auto carve = [&](size_t bytes) -> char* {
        char* p = base + off;
        off += (bytes + 255) & ~(size_t)255;
        return p;
    };
    const size_t BIG = (size_t)Nn * 2048 * sizeof(float);   // 64 MB
    float*    F0 = (float*)carve(BIG);                      // h2 (big hidden)
    float*    F1 = (float*)carve(BIG);                      // xl
    float*    F2 = (float*)carve(BIG);                      // xr
    float*    F3 = (float*)carve(BIG);                      // GAT out / working
    float*    G0 = (float*)carve((size_t)Nn * HID * 4);     // h / h3 [N,512]
    float*    G1 = (float*)carve((size_t)Nn * HID * 4);     // mlp1 out
    float*    LG = (float*)carve((size_t)E * 8 * 4);        // logits
    float*    EX = (float*)carve((size_t)E * 8 * 4);        // exp(logits-max)
    unsigned* LE = (unsigned*)carve((size_t)Nn * 8 * 4);    // encoded seg-max
    float*    LF = (float*)carve((size_t)Nn * 8 * 4);       // decoded seg-max
    float*    DN = (float*)carve((size_t)Nn * 8 * 4);       // softmax denom
    float*    SC = (float*)carve(2048 * 4);                 // BN scale
    float*    SH = (float*)carve(2048 * 4);                 // BN shift
    bf16_t*   B0 = (bf16_t*)carve((size_t)Nn * 2048 * 2);   // bf16 activations
    bf16_t*   B1 = (bf16_t*)carve((size_t)2048 * 2048 * 2); // bf16 weights
    (void)ws_size; (void)n_in; (void)out_size;

    // ===== node projection: h = x @ npW + npB  [N,512] -> G0 =====
    cvt(stream, x, B0, (long long)Nn * 64);
    cvt(stream, npW, B1, 64LL * HID);
    gemm(stream, B0, B1, npB, G0, Nn, HID, 64, 0, 0);

    // ===== conv1: H=8, C=256, hc=2048, concat =====
    cvt(stream, G0, B0, (long long)Nn * HID);          // h (bf16) — reused for res1 too
    cvt(stream, c1Wl, B1, (long long)HID * 2048);
    gemm(stream, B0, B1, c1bl, F1, Nn, 2048, HID, 0, 0);   // xl1
    cvt(stream, c1Wr, B1, (long long)HID * 2048);
    gemm(stream, B0, B1, c1br, F2, Nn, 2048, HID, 0, 0);   // xr1
    fill_f32<<<cdiv((long long)Nn * 2048, 256), 256, 0, stream>>>(F3, 0.f, (long long)Nn * 2048);
    run_gat_edges(stream, F1, F2, ea, c1We, c1at, ei, Nn, E, 8, 256, LG, EX, LE, LF, DN, F3);
    bias_add_rows<<<cdiv((long long)Nn * 2048, 256), 256, 0, stream>>>(F3, c1bs, Nn, 2048);
    // residual: F3 += h @ r1W + r1b  (B0 still holds bf16 h)
    cvt(stream, r1W, B1, (long long)HID * 2048);
    gemm(stream, B0, B1, r1b, F3, Nn, 2048, HID, 0, 1);
    // bn1 + leaky_relu(0.1) -> F0 (= h2, [N,2048])
    bn_stats<<<cdiv(2048, 256), 256, 0, stream>>>(F3, bn1g, bn1b, Nn, 2048, SC, SH);
    bn_apply<<<cdiv((long long)Nn * 2048, 256), 256, 0, stream>>>(F3, F0, SC, SH, Nn, 2048, 1);

    // ===== conv2: H=4, C=512, hc=2048, mean-combine =====
    cvt(stream, F0, B0, (long long)Nn * 2048);         // h2 (bf16) — reused for res2
    cvt(stream, c2Wl, B1, 2048LL * 2048);
    gemm(stream, B0, B1, c2bl, F1, Nn, 2048, 2048, 0, 0);  // xl2
    cvt(stream, c2Wr, B1, 2048LL * 2048);
    gemm(stream, B0, B1, c2br, F2, Nn, 2048, 2048, 0, 0);  // xr2
    fill_f32<<<cdiv((long long)Nn * 2048, 256), 256, 0, stream>>>(F3, 0.f, (long long)Nn * 2048);
    run_gat_edges(stream, F1, F2, ea, c2We, c2at, ei, Nn, E, 4, 512, LG, EX, LE, LF, DN, F3);
    head_mean<<<cdiv((long long)Nn * HID, 256), 256, 0, stream>>>(F3, c2bs, G0, Nn, 4, 512);
    // residual: G0 += h2 @ r2W + r2b
    cvt(stream, r2W, B1, 2048LL * HID);
    gemm(stream, B0, B1, r2b, G0, Nn, HID, 2048, 0, 1);
    // bn2 + elu (in place)
    bn_stats<<<cdiv(HID, 256), 256, 0, stream>>>(G0, bn2g, bn2b, Nn, HID, SC, SH);
    bn_apply<<<cdiv((long long)Nn * HID, 256), 256, 0, stream>>>(G0, G0, SC, SH, Nn, HID, 2);

    // ===== conv3: H=1, C=512, concat =====
    cvt(stream, G0, B0, (long long)Nn * HID);
    cvt(stream, c3Wl, B1, (long long)HID * HID);
    gemm(stream, B0, B1, c3bl, F1, Nn, HID, HID, 0, 0);    // xl3
    cvt(stream, c3Wr, B1, (long long)HID * HID);
    gemm(stream, B0, B1, c3br, F2, Nn, HID, HID, 0, 0);    // xr3
    fill_f32<<<cdiv((long long)Nn * HID, 256), 256, 0, stream>>>(F3, 0.f, (long long)Nn * HID);
    run_gat_edges(stream, F1, F2, ea, c3We, c3at, ei, Nn, E, 1, 512, LG, EX, LE, LF, DN, F3);
    bias_add_rows<<<cdiv((long long)Nn * HID, 256), 256, 0, stream>>>(F3, c3bs, Nn, HID);
    // bn3 + elu (in place)
    bn_stats<<<cdiv(HID, 256), 256, 0, stream>>>(F3, bn3g, bn3b, Nn, HID, SC, SH);
    bn_apply<<<cdiv((long long)Nn * HID, 256), 256, 0, stream>>>(F3, F3, SC, SH, Nn, HID, 2);

    // ===== MLP head =====
    cvt(stream, F3, B0, (long long)Nn * HID);
    cvt(stream, m1W, B1, (long long)HID * HID);
    gemm(stream, B0, B1, m1b, G1, Nn, HID, HID, 0, 0);     // mlp1
    ln_elu<<<Nn, 256, 0, stream>>>(G1, lng, lnb, HID);     // LN + ELU (in place)

    cvt(stream, G1, B0, (long long)Nn * HID);
    cvt(stream, m2W, B1, (long long)HID * 256);
    gemm(stream, B0, B1, m2b, G0, Nn, 256, HID, 1, 0);     // mlp2 + ELU -> [N,256]

    // mlp3 (N=3): tiny VALU kernel, f32 in/out
    mlp3_kernel<<<cdiv((long long)Nn * 3, 256), 256, 0, stream>>>(G0, m3W, m3b, out, Nn);
}
